// SpatialEvoProp_25890062860995
// MI455X (gfx1250) — matrix operand order repaired and checked by reading, not compiled, not run on hardware
//
#include <hip/hip_runtime.h>
#include <math.h>

typedef __attribute__((ext_vector_type(16))) _Float16 v16h;
typedef __attribute__((ext_vector_type(8)))  float    v8f;

#define D64      64
#define K128     128
#define NB       64      // boundaries
#define PROWS    65      // embed table rows
#define TILE_E   64      // edges per tile
#define AST      136     // padded LDS row stride (halves) for A and B panels
#define NUMI     5
#define EDGE_BLOCKS 1024 // persistent blocks for edge kernel

// ---------------- zero ----------------
__global__ void zero_f32(float* __restrict__ p, long n) {
    long i = (long)blockIdx.x * blockDim.x + threadIdx.x;
    long stride = (long)gridDim.x * blockDim.x;
    for (; i < n; i += stride) p[i] = 0.0f;
}

// ---------------- degrees ----------------
__global__ void degree_kernel(const int* __restrict__ src, const int* __restrict__ dst,
                              float* __restrict__ inDeg, float* __restrict__ outDeg, int E) {
    int i = blockIdx.x * blockDim.x + threadIdx.x;
    if (i < E) {
        atomicAdd(&inDeg[dst[i]], 1.0f);
        atomicAdd(&outDeg[src[i]], 1.0f);
    }
}

__global__ void deg_finalize(float* __restrict__ inDeg, float* __restrict__ outDeg, int N) {
    int i = blockIdx.x * blockDim.x + threadIdx.x;
    if (i < N) {
        inDeg[i]  = rsqrtf(fmaxf(inDeg[i],  1.0f));   // d0
        outDeg[i] = rsqrtf(fmaxf(outDeg[i], 1.0f));   // d2
    }
}

// ---------------- proj = embed_table @ G_w.T  (65 x 64) ----------------
__global__ void proj_kernel(const float* __restrict__ embed, const float* __restrict__ Gw,
                            float* __restrict__ proj, int rows) {
    int idx = blockIdx.x * blockDim.x + threadIdx.x;
    int total = rows * D64;
    if (idx < total) {
        int r = idx >> 6, c = idx & 63;
        const float* er = embed + r * 32;
        const float* gr = Gw + c * 32;
        float s = 0.f;
#pragma unroll
        for (int k = 0; k < 32; ++k) s += er[k] * gr[k];
        proj[idx] = s;
    }
}

// ---------------- main edge kernel ----------------
__device__ __forceinline__ int bucket64(const float* __restrict__ bndS, float v) {
    int lo = 0, hi = NB;   // searchsorted(side='left'), result in [0,64]
    while (lo < hi) {
        int mid = (lo + hi) >> 1;
        if (bndS[mid] < v) lo = mid + 1; else hi = mid;
    }
    return lo;
}

__launch_bounds__(256)
__global__ void edge_kernel(const float* __restrict__ feat,
                            const float* __restrict__ loc,
                            const float* __restrict__ proj,
                            const float* __restrict__ aggw,
                            const float* __restrict__ aggb,
                            const float* __restrict__ bnd,
                            const int*   __restrict__ src,
                            const int*   __restrict__ dst,
                            const int*   __restrict__ inter,
                            const float* __restrict__ d0,   // rsqrt in-deg
                            const float* __restrict__ d2,   // rsqrt out-deg
                            float* __restrict__ out,
                            int E, int numTiles) {
    __shared__ float    projS[PROWS * D64];          // 16640 B
    __shared__ _Float16 aggwS[D64 * AST];            // 17408 B  B panel: [n][k]
    __shared__ _Float16 AcatS[TILE_E * AST];         // 17408 B  A panel: [m][k]
    __shared__ float    bndS[NB];
    __shared__ float    aggbS[D64];
    __shared__ float    scaleS[TILE_E];
    __shared__ int      b1S[TILE_E], srcS[TILE_E], dstS[TILE_E];
    __shared__ int      bjS[TILE_E * NUMI], interS[TILE_E * NUMI];

    const int tid  = threadIdx.x;
    const int wv   = tid >> 5;
    const int lane = tid & 31;

    // ---- Phase 0: stage tile-invariant data into LDS (once per block) ----
    for (int i = tid; i < NB; i += 256) bndS[i] = bnd[i];
    for (int i = tid; i < PROWS * D64; i += 256) projS[i] = proj[i];
    for (int i = tid; i < D64 * K128; i += 256) {
        int n = i >> 7, k = i & 127;
        aggwS[n * AST + k] = (_Float16)aggw[i];
    }
    for (int i = tid; i < D64; i += 256) aggbS[i] = aggb[i];
    __syncthreads();

    // ---- wave/WMMA constants ----
    const int mt = wv >> 1;         // M-tile 0..3
    const int nh = wv & 1;          // N-half 0..1
    const int l15   = lane & 15;
    const int lhalf = lane >> 4;    // 0/1
    const int mloc = mt * 16 + l15;
    const int kb = lhalf * 8;       // unused; A/B frags use K-base lhalf*16 within each 32-slice

    // ---- Preload the 8 tile-invariant B fragments into registers ----
    // B is 32x16 (KxN) per fragment: lane<16 -> K 0..15, lane>=16 -> K 16..31, col n = lane&15
    v16h bfr[8];
#pragma unroll
    for (int ks = 0; ks < 4; ++ks) {
#pragma unroll
        for (int nt = 0; nt < 2; ++nt) {
            int n = nh * 32 + nt * 16 + l15;
            const _Float16* br = &aggwS[n * AST + ks * 32 + lhalf * 16];
#pragma unroll
            for (int i = 0; i < 16; ++i) bfr[ks * 2 + nt][i] = br[i];
        }
    }

    const float2* loc2  = (const float2*)loc;
    const float2* feat2 = (const float2*)feat;

    // ---- Persistent loop over edge tiles ----
    for (int t = blockIdx.x; t < numTiles; t += gridDim.x) {

        // ---- Phase A: per-edge scalar work (threads 0..63) ----
        if (tid < TILE_E) {
            int et = t * TILE_E + tid;
            int e  = et < E ? et : (E - 1);
            int s  = src[e], d = dst[e];
            srcS[tid] = s; dstS[tid] = d;
            __builtin_prefetch(feat + (size_t)s * D64, 0, 0);
            float2 ls = loc2[s];
            float2 ld = loc2[d];
            float dx = ld.x - ls.x, dy = ld.y - ls.y;
            b1S[tid] = bucket64(bndS, sqrtf(dx * dx + dy * dy));
#pragma unroll
            for (int j = 0; j < NUMI; ++j) {
                int ij = inter[e * NUMI + j];
                interS[tid * NUMI + j] = ij;
                __builtin_prefetch(feat + (size_t)ij * D64, 0, 0);
                float2 li = loc2[ij];
                float ix = ls.x - li.x, iy = ls.y - li.y;
                bjS[tid * NUMI + j] = bucket64(bndS, sqrtf(ix * ix + iy * iy));
            }
            scaleS[tid] = (et < E) ? d0[d] * d2[s] : 0.0f;
        }
        __syncthreads();

        // ---- Phase B: build A panel (f16) — wave w handles edges 8w..8w+7 ----
#pragma unroll 1
        for (int ee = 0; ee < 8; ++ee) {
            int m = wv * 8 + ee;
            float2 fsrc = feat2[(size_t)srcS[m] * 32 + lane];
            const float* p1 = projS + b1S[m] * D64 + 2 * lane;
            float lo0 = p1[0] * fsrc.x;
            float lo1 = p1[1] * fsrc.y;
            float hx = 0.f, hy = 0.f;
#pragma unroll
            for (int j = 0; j < NUMI; ++j) {
                float2 fi = feat2[(size_t)interS[m * NUMI + j] * 32 + lane];
                const float* pj = projS + bjS[m * NUMI + j] * D64 + 2 * lane;
                hx += pj[0] * fi.x;
                hy += pj[1] * fi.y;
            }
            _Float16* arow = AcatS + m * AST;
            arow[2 * lane]           = (_Float16)lo0;
            arow[2 * lane + 1]       = (_Float16)lo1;
            arow[D64 + 2 * lane]     = (_Float16)(hx * 0.2f);
            arow[D64 + 2 * lane + 1] = (_Float16)(hy * 0.2f);
        }
        __syncthreads();

        // ---- Phase C: WMMA (uniform control flow; EXEC all ones) ----
        v8f acc0 = {};
        v8f acc1 = {};
#pragma unroll
        for (int ks = 0; ks < 4; ++ks) {
            // A fragment (16x32 f16): lane<16 -> K 0..15, lane>=16 -> K 16..31
            const _Float16* ar = &AcatS[mloc * AST + ks * 32 + lhalf * 16];
            v16h a;
#pragma unroll
            for (int i = 0; i < 16; ++i) { a[i] = ar[i]; }
            acc0 = __builtin_amdgcn_wmma_f32_16x16x32_f16(false, a, false, bfr[ks * 2 + 0],
                                                          (short)0, acc0, false, false);
            acc1 = __builtin_amdgcn_wmma_f32_16x16x32_f16(false, a, false, bfr[ks * 2 + 1],
                                                          (short)0, acc1, false, false);
        }

        // ---- Phase D: scale, bias, scatter-add into out[dst] ----
#pragma unroll
        for (int nt = 0; nt < 2; ++nt) {
            v8f cc = nt ? acc1 : acc0;
            int n = nh * 32 + nt * 16 + l15;
            float bias = aggbS[n];
#pragma unroll
            for (int r = 0; r < 8; ++r) {
                int mm = mt * 16 + r + lhalf * 8;   // C layout: VGPR r -> M=r (lanes<16) / M=r+8
                float v = scaleS[mm] * (cc[r] + bias);
                atomicAdd(out + (size_t)dstS[mm] * D64 + n, v);
            }
        }
        __syncthreads();   // protect scaleS/dstS/AcatS before next tile's writes
    }
}

extern "C" void kernel_launch(void* const* d_in, const int* in_sizes, int n_in,
                              void* d_out, int out_size, void* d_ws, size_t ws_size,
                              hipStream_t stream) {
    const float* feat  = (const float*)d_in[0];
    const float* loc   = (const float*)d_in[1];
    const float* embed = (const float*)d_in[2];
    const float* Gw    = (const float*)d_in[3];
    const float* aggw  = (const float*)d_in[4];
    const float* aggb  = (const float*)d_in[5];
    const float* bnd   = (const float*)d_in[6];
    const int*   src   = (const int*)d_in[7];
    const int*   dst   = (const int*)d_in[8];
    const int*   inter = (const int*)d_in[9];

    const int N = in_sizes[0] / D64;
    const int E = in_sizes[7];
    const int prows = in_sizes[2] / 32;   // 65

    float* out   = (float*)d_out;
    float* wsf   = (float*)d_ws;
    float* inDeg = wsf;                   // N floats -> becomes d0
    float* outDeg= wsf + N;               // N floats -> becomes d2
    float* proj  = wsf + 2 * (size_t)N;   // prows*64 floats

    long outN = (long)N * D64;
    zero_f32<<<(int)((outN + 1023) / 1024), 256, 0, stream>>>(out, outN);
    zero_f32<<<(2 * N + 255) / 256, 256, 0, stream>>>(wsf, 2L * N);

    degree_kernel<<<(E + 255) / 256, 256, 0, stream>>>(src, dst, inDeg, outDeg, E);
    deg_finalize<<<(N + 255) / 256, 256, 0, stream>>>(inDeg, outDeg, N);
    proj_kernel<<<(prows * D64 + 255) / 256, 256, 0, stream>>>(embed, Gw, proj, prows);

    const int numTiles = (E + TILE_E - 1) / TILE_E;
    const int blocks = numTiles < EDGE_BLOCKS ? numTiles : EDGE_BLOCKS;
    edge_kernel<<<blocks, 256, 0, stream>>>(
        feat, loc, proj, aggw, aggb, bnd, src, dst, inter,
        inDeg, outDeg, out, E, numTiles);
}
